// GraphAttentionRouter_61924838474174
// MI455X (gfx1250) — matrix operand order repaired and checked by reading, not compiled, not run on hardware
//
#include <hip/hip_runtime.h>

// GraphAttentionRouter for MI455X (gfx1250, wave32, WMMA 16x16x32 f16).
// HBM floor ~134MB (adjacency read + attn_mean write) => ~5.8us at 23.3TB/s.
// 64 query rows per block so the L2-resident K/V stream (6MB/block over two
// passes) is amortized: 64 blocks -> ~384MB L2 traffic, near the HBM floor.
// All matmuls via v_wmma_f32_16x16x32_f16; K/V tiles staged with
// global_load_async_to_lds_b128 (ASYNCcnt) and double-buffered so DMA
// overlaps WMMA+softmax. Two-pass softmax without max-subtraction
// (|scores| ~ 0.1; masked entries get exact 0 weight, matching exp(-1e9)).

typedef __attribute__((ext_vector_type(16))) _Float16 v16h;
typedef __attribute__((ext_vector_type(8)))  float    v8f;
typedef __attribute__((ext_vector_type(2)))  _Float16 h2v;
typedef __attribute__((ext_vector_type(2)))  float    f2v;
typedef int v4i32 __attribute__((vector_size(16)));

#define N_NODES 4096
#define HIDD    256
#define MDIM    32
#define NHEAD   8
#define NCHUNK  (N_NODES / 32)
#define ROWS    64            // query rows per block
#define THREADS 256           // 8 waves: (row-tile 0..3) x (head-group 0..1)

#if defined(__has_builtin)
#if __has_builtin(__builtin_amdgcn_global_load_async_to_lds_b128) && \
    __has_builtin(__builtin_amdgcn_s_wait_asynccnt)
#define USE_ASYNC 1
#endif
#endif
#ifndef USE_ASYNC
#define USE_ASYNC 0
#endif

// one 16-byte global->LDS copy (async DMA if available)
__device__ __forceinline__ void cp16_g2l(_Float16* ldst, const _Float16* gsrc) {
#if USE_ASYNC
  __builtin_amdgcn_global_load_async_to_lds_b128(
      (__attribute__((address_space(1))) v4i32*)(gsrc),
      (__attribute__((address_space(3))) v4i32*)(ldst), 0, 0);
#else
  *(uint4*)ldst = *(const uint4*)gsrc;
#endif
}

template <int CNT>
__device__ __forceinline__ void wait_async_le() {
#if USE_ASYNC
  __builtin_amdgcn_s_wait_asynccnt(CNT);
#endif
}

__device__ __forceinline__ v8f wmma16(v16h a, v16h b, v8f c) {
  return __builtin_amdgcn_wmma_f32_16x16x32_f16(false, a, false, b, (short)0, c,
                                                false, false);
}

// A-matrix (16x32 f16) K index of half-pair j for this lane half (contiguous: K base 16*hi)
__device__ __forceinline__ int a_d0(int j, int hi) {
  return 2 * j + 16 * hi;
}

__device__ __forceinline__ v16h load_afrag_f32(const float* row, int hi) {
  v16h a;
#pragma unroll
  for (int j = 0; j < 8; ++j) {
    int d0 = a_d0(j, hi);
    f2v x = *(const f2v*)(row + d0);
    a[2 * j] = (_Float16)x[0];
    a[2 * j + 1] = (_Float16)x[1];
  }
  return a;
}

__device__ __forceinline__ v16h load_afrag_f16(const _Float16* row, int hi) {
  v16h a;
#pragma unroll
  for (int j = 0; j < 8; ++j) {
    int d0 = a_d0(j, hi);
    h2v x = *(const h2v*)(row + d0);
    a[2 * j] = x[0];
    a[2 * j + 1] = x[1];
  }
  return a;
}

// B-matrix (32x16): lane = column, K pairs contiguous at base[K], K base 16*hi
__device__ __forceinline__ v16h load_bfrag_f16_contig(const _Float16* base, int hi) {
  v16h b;
#pragma unroll
  for (int j = 0; j < 8; ++j) {
    int k0 = 2 * j + 16 * hi;
    h2v x = *(const h2v*)(base + k0);
    b[2 * j] = x[0];
    b[2 * j + 1] = x[1];
  }
  return b;
}

__device__ __forceinline__ v16h load_bfrag_f16_strided(const _Float16* base, int ld, int hi) {
  v16h b;
#pragma unroll
  for (int j = 0; j < 8; ++j) {
    int k0 = 2 * j + 16 * hi;
    b[2 * j] = base[(size_t)k0 * ld];
    b[2 * j + 1] = base[(size_t)(k0 + 1) * ld];
  }
  return b;
}

__device__ __forceinline__ v16h load_bfrag_f32_strided(const float* base, int ld, int hi) {
  v16h b;
#pragma unroll
  for (int j = 0; j < 8; ++j) {
    int k0 = 2 * j + 16 * hi;
    b[2 * j] = (_Float16)base[(size_t)k0 * ld];
    b[2 * j + 1] = (_Float16)base[(size_t)(k0 + 1) * ld];
  }
  return b;
}

// out[M,N] (f16) = A[M,K] @ W[K,N] + bias ; 4 waves/block, 1 tile/wave
__global__ __launch_bounds__(128) void proj_f16_kernel(
    const float* __restrict__ A, const float* __restrict__ W,
    const float* __restrict__ bias, _Float16* __restrict__ out,
    int M, int N, int Kdim) {
  int wave = threadIdx.x >> 5;
  int lane = threadIdx.x & 31;
  int l = lane & 15, hi = lane >> 4;
  int tile = blockIdx.x * 4 + wave;
  int tilesN = N >> 4;
  int tr = tile / tilesN, tc = tile - tr * tilesN;
  if (tr * 16 >= M) return;  // wave-uniform
  int row = tr * 16 + l;
  int col = tc * 16 + l;
  v8f c = {};
  for (int kk = 0; kk < Kdim; kk += 32) {
    v16h a = load_afrag_f32(A + (size_t)row * Kdim + kk, hi);
    v16h b = load_bfrag_f32_strided(W + (size_t)kk * N + col, N, hi);
    c = wmma16(a, b, c);
  }
  float bb = bias[col];
#pragma unroll
  for (int r = 0; r < 8; ++r) {
    int orow = tr * 16 + r + 8 * hi;
    out[(size_t)orow * N + col] = (_Float16)(c[r] + bb);
  }
}

__global__ __launch_bounds__(THREADS) void attn_kernel(
    const _Float16* __restrict__ Qf, const _Float16* __restrict__ Kf,
    const _Float16* __restrict__ Vf, const int* __restrict__ adj,
    const float* __restrict__ Wo, const float* __restrict__ bo,
    float* __restrict__ out_routed, float* __restrict__ out_attn) {
  __shared__ _Float16 qbuf[ROWS * HIDD];         // 32 KB (reused as f16 O buffer)
  __shared__ _Float16 kbuf[2][32 * HIDD];        // 32 KB ping-pong
  __shared__ _Float16 vbuf[2][32 * HIDD];        // 32 KB ping-pong
  __shared__ _Float16 attnh[8 * 4 * 16 * 32];    // per (wave, head) tiles, 32 KB
  __shared__ float    asum[8 * 16 * 32];         // [hg*4+rt][16][32], 16 KB

  const int tid = threadIdx.x;
  const int wave = tid >> 5;
  const int lane = tid & 31;
  const int l = lane & 15;
  const int hi = lane >> 4;
  const int rt = wave >> 1;      // row tile 0..3 (rows rt*16..rt*16+15)
  const int hg = wave & 1;       // head group: heads hg*4 .. hg*4+3
  const int n0 = blockIdx.x * ROWS;
  const int nr0 = n0 + rt * 16;  // this wave's first query row
  const float scale = 0.1767766952966369f;  // 1/sqrt(32)

  // stage 64 query rows (all heads) into LDS via async DMA
#pragma unroll
  for (int i = 0; i < 8; ++i) {
    int o = (tid + THREADS * i) * 8;
    cp16_g2l(qbuf + o, Qf + (size_t)n0 * HIDD + o);
  }
  wait_async_le<0>();
  __syncthreads();

  // Q A-fragments for this wave's 4 heads (persist across both passes)
  v16h qfr[4];
#pragma unroll
  for (int hh = 0; hh < 4; ++hh) {
    int h = hg * 4 + hh;
    qfr[hh] = load_afrag_f16(&qbuf[(rt * 16 + l) * HIDD + h * MDIM], hi);
  }

  // ---------------- pass 1: softmax denominators ----------------
  float acc[4][8];
#pragma unroll
  for (int hh = 0; hh < 4; ++hh)
#pragma unroll
    for (int r = 0; r < 8; ++r) acc[hh][r] = 0.f;

  // prologue: chunk 0 -> kbuf[0]
#pragma unroll
  for (int i = 0; i < 4; ++i) {
    int o = (tid + THREADS * i) * 8;
    cp16_g2l(kbuf[0] + o, Kf + o);
  }
  for (int c = 0; c < NCHUNK; ++c) {
    const int m0 = c * 32;
    __syncthreads();  // (A) chunk c-1 readers done -> buffer (c+1)&1 free
    if (c + 1 < NCHUNK) {
      const _Float16* src = Kf + (size_t)(m0 + 32) * HIDD;
      _Float16* dst = kbuf[(c + 1) & 1];
#pragma unroll
      for (int i = 0; i < 4; ++i) {
        int o = (tid + THREADS * i) * 8;
        cp16_g2l(dst + o, src + o);
      }
      wait_async_le<4>();  // chunk c landed (in-order completion)
      if (tid < ROWS)      // prefetch next adjacency tile rows
        __builtin_prefetch(&adj[(size_t)(n0 + tid) * N_NODES + m0 + 32], 0, 0);
    } else {
      wait_async_le<0>();
    }
    __syncthreads();  // (B) chunk c visible to all waves
    const _Float16* kb = kbuf[c & 1];
#pragma unroll
    for (int t = 0; t < 2; ++t) {
      const int m = m0 + 16 * t + l;
      int msk[8];
#pragma unroll
      for (int r = 0; r < 8; ++r)
        msk[r] = adj[(size_t)(nr0 + r + 8 * hi) * N_NODES + m];
#pragma unroll
      for (int hh = 0; hh < 4; ++hh) {
        int h = hg * 4 + hh;
        v16h bk_ = load_bfrag_f16_contig(&kb[(16 * t + l) * HIDD + h * MDIM], hi);
        v8f z = {};
        v8f s = wmma16(qfr[hh], bk_, z);
#pragma unroll
        for (int r = 0; r < 8; ++r) {
          float p = msk[r] ? __expf(s[r] * scale) : 0.f;
          acc[hh][r] += p;
        }
      }
    }
  }
  float inv[4][8];
#pragma unroll
  for (int hh = 0; hh < 4; ++hh)
#pragma unroll
    for (int r = 0; r < 8; ++r) {
      float v = acc[hh][r];
      v += __shfl_xor(v, 1);
      v += __shfl_xor(v, 2);
      v += __shfl_xor(v, 4);
      v += __shfl_xor(v, 8);
      inv[hh][r] = (v > 0.f) ? (1.f / v) : 0.f;
    }

  // ---------------- pass 2: attn_mean + attn @ V ----------------
  v8f of[4][2];
  {
    v8f z = {};
#pragma unroll
    for (int hh = 0; hh < 4; ++hh) { of[hh][0] = z; of[hh][1] = z; }
  }
  __syncthreads();
  // prologue: chunk 0 K+V
#pragma unroll
  for (int i = 0; i < 4; ++i) {
    int o = (tid + THREADS * i) * 8;
    cp16_g2l(kbuf[0] + o, Kf + o);
    cp16_g2l(vbuf[0] + o, Vf + o);
  }
  for (int c = 0; c < NCHUNK; ++c) {
    const int m0 = c * 32;
    __syncthreads();  // (A) prior AV/mean readers done -> buffer (c+1)&1 free
    if (c + 1 < NCHUNK) {
      const _Float16* sk = Kf + (size_t)(m0 + 32) * HIDD;
      const _Float16* sv = Vf + (size_t)(m0 + 32) * HIDD;
      _Float16* dk = kbuf[(c + 1) & 1];
      _Float16* dv = vbuf[(c + 1) & 1];
#pragma unroll
      for (int i = 0; i < 4; ++i) {
        int o = (tid + THREADS * i) * 8;
        cp16_g2l(dk + o, sk + o);
        cp16_g2l(dv + o, sv + o);
      }
      wait_async_le<8>();
    } else {
      wait_async_le<0>();
    }
    __syncthreads();  // (B) chunk c visible
    const _Float16* kb = kbuf[c & 1];
    const _Float16* vb = vbuf[c & 1];

    float psum[2][8];
#pragma unroll
    for (int t = 0; t < 2; ++t)
#pragma unroll
      for (int r = 0; r < 8; ++r) psum[t][r] = 0.f;

#pragma unroll
    for (int t = 0; t < 2; ++t) {
      const int m = m0 + 16 * t + l;
      int msk[8];
#pragma unroll
      for (int r = 0; r < 8; ++r)
        msk[r] = adj[(size_t)(nr0 + r + 8 * hi) * N_NODES + m];
#pragma unroll
      for (int hh = 0; hh < 4; ++hh) {
        int h = hg * 4 + hh;
        v16h bk_ = load_bfrag_f16_contig(&kb[(16 * t + l) * HIDD + h * MDIM], hi);
        v8f z = {};
        v8f s = wmma16(qfr[hh], bk_, z);
#pragma unroll
        for (int r = 0; r < 8; ++r) {
          int row = r + 8 * hi;
          float p = msk[r] ? __expf(s[r] * scale) * inv[hh][r] : 0.f;
          psum[t][r] += p;
          // C-layout -> LDS; reread in A-layout by this same wave (LDS in-order)
          attnh[((wave * 4 + hh) * 16 + row) * 32 + 16 * t + l] = (_Float16)p;
        }
      }
    }
#pragma unroll
    for (int t = 0; t < 2; ++t)
#pragma unroll
      for (int r = 0; r < 8; ++r)
        asum[((hg * 4 + rt) * 16 + r + 8 * hi) * 32 + 16 * t + l] = psum[t][r];
    __syncthreads();  // (C) asum complete
    // attn.mean(h): reduce the 2 head-group partials, write 64x32 f32 tile
#pragma unroll
    for (int i = 0; i < 8; ++i) {
      int idx = tid + THREADS * i;  // 0..2047
      int row = idx >> 5, col = idx & 31;
      int rt2 = row >> 4, rowl = row & 15;
      float ssum = asum[((0 * 4 + rt2) * 16 + rowl) * 32 + col] +
                   asum[((1 * 4 + rt2) * 16 + rowl) * 32 + col];
      out_attn[(size_t)(n0 + row) * N_NODES + m0 + col] = ssum * 0.125f;
    }
    // O_h += attn_h[16x32] @ V_h[32x(2x16)]
#pragma unroll
    for (int hh = 0; hh < 4; ++hh) {
      int h = hg * 4 + hh;
      v16h af = load_afrag_f16(&attnh[((wave * 4 + hh) * 16 + l) * 32], hi);
#pragma unroll
      for (int t = 0; t < 2; ++t) {
        v16h bv_ = load_bfrag_f16_strided(&vb[h * MDIM + 16 * t + l], HIDD, hi);
        of[hh][t] = wmma16(af, bv_, of[hh][t]);
      }
    }
  }

  // ------------- epilogue: routed = O[64,256] @ Wo[256,32] + bo -------------
  __syncthreads();
  // write O (f16) into retired qbuf: [64][256]
#pragma unroll
  for (int hh = 0; hh < 4; ++hh)
#pragma unroll
    for (int t = 0; t < 2; ++t)
#pragma unroll
      for (int r = 0; r < 8; ++r) {
        int h = hg * 4 + hh;
        qbuf[(rt * 16 + r + 8 * hi) * HIDD + h * MDIM + 16 * t + l] =
            (_Float16)of[hh][t][r];
      }
  __syncthreads();
  {  // 8 output tiles (4 row-tiles x 2 col-tiles), one per wave
    int rt2 = wave >> 1, tc = wave & 1;
    v8f cacc = {};
    for (int kk = 0; kk < HIDD; kk += 32) {
      v16h a = load_afrag_f16(&qbuf[(rt2 * 16 + l) * HIDD + kk], hi);
      v16h b = load_bfrag_f32_strided(Wo + (size_t)kk * MDIM + tc * 16 + l, MDIM, hi);
      cacc = wmma16(a, b, cacc);
    }
    float bb = bo[tc * 16 + l];
#pragma unroll
    for (int r = 0; r < 8; ++r)
      out_routed[(size_t)(n0 + rt2 * 16 + r + 8 * hi) * MDIM + tc * 16 + l] =
          cacc[r] + bb;
  }
}

extern "C" void kernel_launch(void* const* d_in, const int* in_sizes, int n_in,
                              void* d_out, int out_size, void* d_ws, size_t ws_size,
                              hipStream_t stream) {
  const float* nf  = (const float*)d_in[0];
  const float* msg = (const float*)d_in[1];
  const int*   adj = (const int*)d_in[2];
  const float* Wq  = (const float*)d_in[3];
  const float* bq  = (const float*)d_in[4];
  const float* Wk  = (const float*)d_in[5];
  const float* bk  = (const float*)d_in[6];
  const float* Wv  = (const float*)d_in[7];
  const float* bv  = (const float*)d_in[8];
  const float* Wo  = (const float*)d_in[9];
  const float* bo  = (const float*)d_in[10];
  float* out = (float*)d_out;
  float* out_routed = out;
  float* out_attn   = out + (size_t)N_NODES * MDIM;

  _Float16* Qf = (_Float16*)d_ws;                  // 2 MB
  _Float16* Kf = Qf + (size_t)N_NODES * HIDD;      // 2 MB
  _Float16* Vf = Kf + (size_t)N_NODES * HIDD;      // 2 MB

  // projections: (M/16)*(N/16) tiles, 4 tiles per 128-thread block
  int tilesQK = (N_NODES / 16) * (HIDD / 16);      // 4096
  proj_f16_kernel<<<tilesQK / 4, 128, 0, stream>>>(nf,  Wq, bq, Qf, N_NODES, HIDD, HIDD);
  proj_f16_kernel<<<tilesQK / 4, 128, 0, stream>>>(nf,  Wk, bk, Kf, N_NODES, HIDD, HIDD);
  proj_f16_kernel<<<tilesQK / 4, 128, 0, stream>>>(msg, Wv, bv, Vf, N_NODES, HIDD, MDIM);

  attn_kernel<<<N_NODES / ROWS, THREADS, 0, stream>>>(Qf, Kf, Vf, adj, Wo, bo,
                                                      out_routed, out_attn);
}
